// gcn_rnn_diagnoser_62852551410032
// MI455X (gfx1250) — compile-verified, matrix operand order checked by reading, not run on hardware
//
#include <hip/hip_runtime.h>
#include <math.h>

typedef __attribute__((ext_vector_type(16))) _Float16 v16h;
typedef __attribute__((ext_vector_type(8)))  _Float16 v8h;
typedef __attribute__((ext_vector_type(8)))  float    v8f;

#define NODE    5
#define TSTEPS  512
#define BT      16          // batch rows per workgroup (one WMMA M-tile)
#define STRIDE  648         // LDS row stride in halves (bank-conflict-free b128 reads)

// folded GEMM shapes (all K padded to multiples of 32; N multiples of 16)
// L0: K=352 (325 padded), N=640 ; L1: K=640, N=640 ; L2: K=640, N=320
#define KT0 11
#define NT0 40
#define KT1 20
#define NT1 40
#define KT2 20
#define NT2 20
#define NB0 (NT0*KT0)
#define NB1 (NT1*KT1)
#define NB2 (NT2*KT2)
#define WELEM0 (NB0*512)
#define WELEM1 (NB1*512)
#define WELEM2 (NB2*512)
#define WTOT   (WELEM0+WELEM1+WELEM2)

union Frag { v16h v; v8h h[2]; };

// K index held by (lane, half) for 16-bit A(16x32) / B(32x16) fragments (ISA 7.12.2)
__device__ __forceinline__ int kmap(int lane, int h) {
  int base = (lane < 16) ? 0 : 8;
  return (h < 8) ? (base + h) : (16 + base + (h - 8));
}

// ---------------------------------------------------------------------------
// prep: fold adjacency into weights, pre-swizzle to B-fragment layout (f16),
//       replicate biases, zero BN accumulators
// ---------------------------------------------------------------------------
__global__ __launch_bounds__(256) void prep_kernel(
    const float* __restrict__ adj,
    const float* __restrict__ W0, const float* __restrict__ b0,
    const float* __restrict__ W1, const float* __restrict__ b1,
    const float* __restrict__ W2, const float* __restrict__ b2,
    _Float16* __restrict__ wb0, _Float16* __restrict__ wb1, _Float16* __restrict__ wb2,
    float* __restrict__ b0r, float* __restrict__ b1r, float* __restrict__ b2r,
    float* __restrict__ sums)
{
  int idx = blockIdx.x * 256 + threadIdx.x;
  // A = adj.T  =>  A[i][j] = adj[j*NODE + i]
  if (idx < WELEM0) {
    int e = idx;
    int block = e >> 9, inb = e & 511, lane = inb >> 4, h = inb & 15;
    int nt = block / KT0, kt = block % KT0;
    int n = nt * 16 + (lane & 15);
    int k = kt * 32 + kmap(lane, h);
    float v = 0.f;
    if (k < 325) {
      int j = k / 65, kk = k % 65;
      int i = n >> 7, f = n & 127;
      v = adj[j * NODE + i] * W0[kk * 128 + f];
    }
    wb0[e] = (_Float16)v;
  } else if (idx < WELEM0 + WELEM1) {
    int e = idx - WELEM0;
    int block = e >> 9, inb = e & 511, lane = inb >> 4, h = inb & 15;
    int nt = block / KT1, kt = block % KT1;
    int n = nt * 16 + (lane & 15);
    int k = kt * 32 + kmap(lane, h);
    int j = k >> 7, kk = k & 127;
    int i = n >> 7, f = n & 127;
    wb1[e] = (_Float16)(adj[j * NODE + i] * W1[kk * 128 + f]);
  } else if (idx < WTOT) {
    int e = idx - WELEM0 - WELEM1;
    int block = e >> 9, inb = e & 511, lane = inb >> 4, h = inb & 15;
    int nt = block / KT2, kt = block % KT2;
    int n = nt * 16 + (lane & 15);
    int k = kt * 32 + kmap(lane, h);
    int j = k >> 7, kk = k & 127;
    int i = n >> 6, f = n & 63;
    wb2[e] = (_Float16)(adj[j * NODE + i] * W2[kk * 64 + f]);
  } else {
    int r = idx - WTOT;
    if (r < 640)              b0r[r]        = b0[r & 127];
    else if (r < 1280)        b1r[r - 640]  = b1[(r - 640) & 127];
    else if (r < 1600)        b2r[r - 1280] = b2[(r - 1280) & 63];
    else if (r < 1600 + 512)  sums[r - 1600] = 0.f;   // sum[256] + sumsq[256]
  }
}

// ---------------------------------------------------------------------------
// WMMA GEMM over one 16-row M tile resident in LDS
// ACT: 0 = relu, 1 = tanh
// ---------------------------------------------------------------------------
template <int KT, int NT, int ACT>
__device__ __forceinline__ void gemm_step(
    const _Float16* __restrict__ src, const _Float16* __restrict__ W,
    const float* __restrict__ bias, _Float16* __restrict__ dst,
    int lane, int wave)
{
  const int lrow  = (lane < 16) ? lane : (lane - 16);
  const int koff0 = (lane < 16) ? 0 : 8;
  for (int nt = wave; nt < NT; nt += 8) {
    float bv = bias[nt * 16 + (lane & 15)];
    v8f acc;
#pragma unroll
    for (int m = 0; m < 8; ++m) acc[m] = bv;
    const _Float16* wp = W + (((size_t)(nt * KT)) << 9) + ((size_t)lane << 4);
    for (int kt = 0; kt < KT; ++kt) {
      Frag a, b;
      const _Float16* ap = src + lrow * STRIDE + kt * 32 + koff0;
      a.h[0] = *(const v8h*)(ap);
      a.h[1] = *(const v8h*)(ap + 16);
      b.h[0] = *(const v8h*)(wp);
      b.h[1] = *(const v8h*)(wp + 8);
      wp += 512;
      acc = __builtin_amdgcn_wmma_f32_16x16x32_f16(
          false, a.v, false, b.v, (short)0, acc, false, false);
    }
    int col = nt * 16 + (lane & 15);
    int rb  = (lane >> 4) * 8;
#pragma unroll
    for (int m = 0; m < 8; ++m) {
      float vv = acc[m];
      vv = (ACT == 0) ? fmaxf(vv, 0.0f) : tanhf(vv);
      dst[(rb + m) * STRIDE + col] = (_Float16)vv;
    }
  }
}

// ---------------------------------------------------------------------------
// recur: 512-step recurrence, state in LDS, then fc1 + BN partial sums.
// x_t tiles are double-buffered into LDS with CDNA5 async loads
// (GLOBAL_LOAD_ASYNC_TO_LDS_B32, tracked by ASYNCcnt).
// ---------------------------------------------------------------------------
__global__ __launch_bounds__(256) void recur_kernel(
    const float* __restrict__ x,
    const _Float16* __restrict__ wb0, const _Float16* __restrict__ wb1,
    const _Float16* __restrict__ wb2,
    const float* __restrict__ b0r, const float* __restrict__ b1r,
    const float* __restrict__ b2r,
    const float* __restrict__ fc1_w, const float* __restrict__ fc1_b,
    float* __restrict__ h1, float* __restrict__ sum, float* __restrict__ sumsq)
{
  __shared__ __align__(16) _Float16 bufA[BT * STRIDE];
  __shared__ __align__(16) _Float16 bufB[BT * STRIDE];
  __shared__ __align__(16) float xstage[2][BT * NODE];   // async-staged x_t
  __shared__ float red[512];

  const int tid   = threadIdx.x;
  const int lane  = tid & 31;
  const int wave  = tid >> 5;
  const int bbase = blockIdx.x * BT;

  // state (bufB cols 0..319) starts at zero
  for (int i = tid; i < BT * STRIDE; i += 256) bufB[i] = (_Float16)0.f;

  // prologue: async-stage x_{t=0} into xstage[0]
  if (tid < BT * NODE) {
    const float* gp = &x[((size_t)(bbase + tid / NODE) * NODE + (tid % NODE)) * TSTEPS];
    unsigned ldsa = (unsigned)(uintptr_t)&xstage[0][tid];
    asm volatile("global_load_async_to_lds_b32 %0, %1, off"
                 :: "v"(ldsa), "v"(gp) : "memory");
  }

  for (int t = 0; t < TSTEPS; ++t) {
    // (a) staged x for this step is in LDS once ASYNCcnt drains
    asm volatile("s_wait_asynccnt 0x0" ::: "memory");
    __syncthreads();

    const float* xcur = xstage[t & 1];
    // build Hin in bufA cols [0,352): per node j, col j*65 = x_t, cols +1..+64 = state
    for (int idx = tid; idx < BT * 352; idx += 256) {
      int r = idx / 352, c = idx % 352;
      float v = 0.f;
      if (c < 325) {
        int j = c / 65, cc = c % 65;
        if (cc == 0) v = xcur[r * NODE + j];
        else         v = (float)bufB[r * STRIDE + j * 64 + (cc - 1)];
      }
      bufA[r * STRIDE + c] = (_Float16)v;
    }

    // kick off async stage of x_{t+1} into the other buffer (overlaps GEMMs)
    if (t + 1 < TSTEPS && tid < BT * NODE) {
      const float* gp =
          &x[((size_t)(bbase + tid / NODE) * NODE + (tid % NODE)) * TSTEPS + t + 1];
      unsigned ldsa = (unsigned)(uintptr_t)&xstage[(t + 1) & 1][tid];
      asm volatile("global_load_async_to_lds_b32 %0, %1, off"
                   :: "v"(ldsa), "v"(gp) : "memory");
    }
    __syncthreads();
    gemm_step<KT0, NT0, 0>(bufA, wb0, b0r, bufB, lane, wave);  // relu -> bufB[.,640]
    __syncthreads();
    gemm_step<KT1, NT1, 0>(bufB, wb1, b1r, bufA, lane, wave);  // relu -> bufA[.,640]
    __syncthreads();
    gemm_step<KT2, NT2, 1>(bufA, wb2, b2r, bufB, lane, wave);  // tanh -> state bufB[.,320]
    // no trailing barrier: top-of-loop barrier (a) orders G3 output vs next build
  }
  __syncthreads();

  // ---- fc1: xs = [x_end(5) | state(320)] @ fc1_w(325x256) + b, relu ----
  const int r  = tid >> 4;   // 0..15
  const int cg = tid & 15;   // 16-col group
  const float* xlast = xstage[(TSTEPS - 1) & 1];

  float acc[16];
#pragma unroll
  for (int c = 0; c < 16; ++c) acc[c] = fc1_b[cg * 16 + c];
#pragma unroll
  for (int j = 0; j < NODE; ++j) {
    float a = xlast[r * NODE + j];
    const float* wr = fc1_w + j * 256 + cg * 16;
#pragma unroll
    for (int c = 0; c < 16; ++c) acc[c] += a * wr[c];
  }
  for (int k = 0; k < 320; ++k) {
    float a = (float)bufB[r * STRIDE + k];
    const float* wr = fc1_w + (NODE + k) * 256 + cg * 16;
#pragma unroll
    for (int c = 0; c < 16; ++c) acc[c] += a * wr[c];
  }
#pragma unroll
  for (int c = 0; c < 16; ++c) acc[c] = fmaxf(acc[c], 0.f);

  float* hrow = h1 + (size_t)(bbase + r) * 256 + cg * 16;
#pragma unroll
  for (int c = 0; c < 16; ++c) hrow[c] = acc[c];

  // ---- BN partial sums: LDS reduce then one global atomic per column ----
  for (int i = tid; i < 512; i += 256) red[i] = 0.f;
  __syncthreads();
#pragma unroll
  for (int c = 0; c < 16; ++c) {
    atomicAdd(&red[cg * 16 + c], acc[c]);
    atomicAdd(&red[256 + cg * 16 + c], acc[c] * acc[c]);
  }
  __syncthreads();
  atomicAdd(&sum[tid],   red[tid]);
  atomicAdd(&sumsq[tid], red[256 + tid]);
}

// ---------------------------------------------------------------------------
// head: batchnorm (batch stats) + fc2 + softmax
// ---------------------------------------------------------------------------
__global__ __launch_bounds__(256) void head_kernel(
    const float* __restrict__ h1, const float* __restrict__ sum,
    const float* __restrict__ sumsq,
    const float* __restrict__ gamma, const float* __restrict__ beta,
    const float* __restrict__ fc2_w, const float* __restrict__ fc2_b,
    float* __restrict__ out, int B)
{
  __shared__ float smean[256], srstd[256], sgam[256], sbet[256];
  int tid = threadIdx.x;
  float invB = 1.0f / (float)B;
  float m = sum[tid] * invB;
  float v = sumsq[tid] * invB - m * m;
  smean[tid] = m;
  srstd[tid] = rsqrtf(v + 1e-5f);
  sgam[tid]  = gamma[tid];
  sbet[tid]  = beta[tid];
  __syncthreads();

  int row = blockIdx.x * 256 + tid;
  float logit[7];
#pragma unroll
  for (int c = 0; c < 7; ++c) logit[c] = fc2_b[c];
  const float* hrow = h1 + (size_t)row * 256;
  for (int k = 0; k < 256; ++k) {
    float h = (hrow[k] - smean[k]) * srstd[k] * sgam[k] + sbet[k];
    const float* wr = fc2_w + k * 7;
#pragma unroll
    for (int c = 0; c < 7; ++c) logit[c] += h * wr[c];
  }
  float mx = logit[0];
#pragma unroll
  for (int c = 1; c < 7; ++c) mx = fmaxf(mx, logit[c]);
  float s = 0.f;
#pragma unroll
  for (int c = 0; c < 7; ++c) { logit[c] = expf(logit[c] - mx); s += logit[c]; }
  float inv = 1.0f / s;
#pragma unroll
  for (int c = 0; c < 7; ++c) out[(size_t)row * 7 + c] = logit[c] * inv;
}

// ---------------------------------------------------------------------------
extern "C" void kernel_launch(void* const* d_in, const int* in_sizes, int n_in,
                              void* d_out, int out_size, void* d_ws, size_t ws_size,
                              hipStream_t stream) {
  const float* x     = (const float*)d_in[0];
  const float* adj   = (const float*)d_in[1];
  const float* W0    = (const float*)d_in[2];
  const float* b0    = (const float*)d_in[3];
  const float* W1    = (const float*)d_in[4];
  const float* b1    = (const float*)d_in[5];
  const float* W2    = (const float*)d_in[6];
  const float* b2    = (const float*)d_in[7];
  const float* fc1_w = (const float*)d_in[8];
  const float* fc1_b = (const float*)d_in[9];
  const float* gamma = (const float*)d_in[10];
  const float* beta  = (const float*)d_in[11];
  const float* fc2_w = (const float*)d_in[12];
  const float* fc2_b = (const float*)d_in[13];

  const int B = in_sizes[0] / (NODE * TSTEPS);   // 8192

  // workspace layout
  char* ws = (char*)d_ws;
  _Float16* wb0 = (_Float16*)ws;
  _Float16* wb1 = wb0 + WELEM0;
  _Float16* wb2 = wb1 + WELEM1;
  float* fbase  = (float*)(ws + (size_t)2 * WTOT);  // 16B aligned
  float* b0r    = fbase;                            // 640
  float* b1r    = b0r + 640;                        // 640
  float* b2r    = b1r + 640;                        // 320
  float* sums   = b2r + 320;                        // 512 (sum|sumsq)
  float* h1     = sums + 512;                       // B*256

  // 1) fold + swizzle weights, biases, zero BN accumulators
  int prep_total = WTOT + 1600 + 512;
  prep_kernel<<<(prep_total + 255) / 256, 256, 0, stream>>>(
      adj, W0, b0, W1, b1, W2, b2, wb0, wb1, wb2, b0r, b1r, b2r, sums);

  // 2) recurrence (WMMA + async-to-LDS x staging) + fc1 + BN partials
  recur_kernel<<<B / BT, 256, 0, stream>>>(
      x, wb0, wb1, wb2, b0r, b1r, b2r, fc1_w, fc1_b, h1, sums, sums + 256);

  // 3) BN + fc2 + softmax
  head_kernel<<<B / 256, 256, 0, stream>>>(
      h1, sums, sums + 256, gamma, beta, fc2_w, fc2_b, (float*)d_out, B);
}
